// MambaSequenceClassifier_68564857913646
// MI455X (gfx1250) — compile-verified, matrix operand order checked by reading, not run on hardware
//
#include <hip/hip_runtime.h>
#include <hip/hip_bf16.h>
#include <math.h>

// ---------------- problem constants ----------------
#define BATCH   4
#define SEQ     512
#define NTOK    (BATCH * SEQ)          // 2048 tokens
#define D_MODEL 2048
#define D_INNER 4096
#define D_STATE 64
#define D_CONV  4
#define DT_RANK 128
#define HID     1364
#define NCLS    16

// ---------------- WMMA types ----------------
typedef __attribute__((ext_vector_type(16))) __bf16 v16bf;
typedef __attribute__((ext_vector_type(2)))  __bf16 v2bf;
typedef __attribute__((ext_vector_type(8)))  float  v8f;

union BF16Frag { v16bf v; __bf16 e[16]; };
union AccFrag  { v8f   v; float  f[8]; };
union V4BF     { __bf16 e[4]; unsigned long long u; };

static __device__ __forceinline__ __bf16 to_bf16(float x) { return (__bf16)x; }
static __device__ __forceinline__ float silu(float x) { return x / (1.0f + __expf(-x)); }

// ===================================================================
// GEMM:  C[m, c_off+n] = act( sum_k A[m,k] * W[n,k] + bias[n] )
//   A: [M,K] row-major f32,  W: [N,K] row-major f32 (weight layout).
// Block tile: 128(M) x 64(N) x 64(K), 256 threads = 8 waves.
// Waves: 4(M-groups of 32) x 2(N-groups of 32); wave tile 32x32 = 2x2 WMMA.
// Software-pipelined: global loads for tile k+1 issue before the WMMAs of
// tile k, so HBM latency hides behind matrix work.
// Requirements (hold for all call sites): M % 128 == 0, K % 64 == 0.
// Only N may be ragged (handled branchlessly).
// ACT: 0 = none, 1 = softplus
// ===================================================================
template <int ACT>
__global__ __launch_bounds__(256) void gemm_bf16_wmma(
    const float* __restrict__ A, int lda,
    const float* __restrict__ W, int ldw,
    const float* __restrict__ bias,
    float* __restrict__ C, int ldc, int c_off,
    int M, int N, int K)
{
    __shared__ __bf16 As[128][64];   // 16 KB
    __shared__ __bf16 Bs[64][64];    //  8 KB

    const int tid  = threadIdx.x;
    const int lane = tid & 31;
    const int wave = tid >> 5;         // 0..7
    const int wm   = wave >> 1;        // 0..3 -> M offset 0/32/64/96
    const int wn   = wave & 1;         // 0..1 -> N offset 0/32
    const int lm   = lane & 15;
    const int half = lane >> 4;        // 0/1

    const int m0 = blockIdx.y * 128;
    const int n0 = blockIdx.x * 64;

    // staging coordinates: 16 threads per row-slice, 4 floats each
    const int sr = tid >> 4;           // 0..15
    const int sc = (tid & 15) << 2;    // 0,4,...,60

    // loop-invariant staging row pointers + ragged-N predicate for B
    const float* aptr[8];
    const float* bptr[4];
    bool bvalid[4];
    #pragma unroll
    for (int p = 0; p < 8; ++p)
        aptr[p] = A + (size_t)(m0 + sr + p * 16) * lda + sc;
    #pragma unroll
    for (int p = 0; p < 4; ++p) {
        int gn = n0 + sr + p * 16;
        bvalid[p] = gn < N;
        bptr[p] = W + (size_t)(bvalid[p] ? gn : N - 1) * ldw + sc;
    }

    v8f acc[2][2];
    #pragma unroll
    for (int mi = 0; mi < 2; ++mi)
        #pragma unroll
        for (int ni = 0; ni < 2; ++ni)
            acc[mi][ni] = (v8f){};

    // ---- prologue: load K-tile 0 into registers
    float4 ra[8], rb[4];
    #pragma unroll
    for (int p = 0; p < 8; ++p) ra[p] = *(const float4*)aptr[p];
    #pragma unroll
    for (int p = 0; p < 4; ++p) rb[p] = *(const float4*)bptr[p];

    for (int k0 = 0; k0 < K; k0 += 64) {
        // ---- convert staged registers -> LDS (one batched loadcnt wait)
        #pragma unroll
        for (int p = 0; p < 8; ++p) {
            V4BF t;
            t.e[0] = to_bf16(ra[p].x); t.e[1] = to_bf16(ra[p].y);
            t.e[2] = to_bf16(ra[p].z); t.e[3] = to_bf16(ra[p].w);
            *(unsigned long long*)&As[sr + p * 16][sc] = t.u;
        }
        #pragma unroll
        for (int p = 0; p < 4; ++p) {
            float4 v = rb[p];
            if (!bvalid[p]) { v.x = 0.f; v.y = 0.f; v.z = 0.f; v.w = 0.f; }
            V4BF t;
            t.e[0] = to_bf16(v.x); t.e[1] = to_bf16(v.y);
            t.e[2] = to_bf16(v.z); t.e[3] = to_bf16(v.w);
            *(unsigned long long*)&Bs[sr + p * 16][sc] = t.u;
        }
        __syncthreads();

        // ---- issue global loads for the NEXT K tile (no wait here)
        if (k0 + 64 < K) {
            #pragma unroll
            for (int p = 0; p < 8; ++p) ra[p] = *(const float4*)(aptr[p] + k0 + 64);
            #pragma unroll
            for (int p = 0; p < 4; ++p) rb[p] = *(const float4*)(bptr[p] + k0 + 64);
            if (k0 + 128 < K)
                __builtin_prefetch(aptr[0] + k0 + 128, 0, 1);
        }

        // ---- two K=32 WMMA steps over the staged K=64 slice
        #pragma unroll
        for (int kk = 0; kk < 64; kk += 32) {
            BF16Frag fa[2], fb[2];
            #pragma unroll
            for (int v = 0; v < 8; ++v) {
                int k = kk + ((v >> 2) << 4) + (half << 3) + ((v & 3) << 1);
                v2bf pa0 = *(const v2bf*)&As[wm * 32 + lm][k];
                v2bf pa1 = *(const v2bf*)&As[wm * 32 + 16 + lm][k];
                v2bf pb0 = *(const v2bf*)&Bs[wn * 32 + lm][k];
                v2bf pb1 = *(const v2bf*)&Bs[wn * 32 + 16 + lm][k];
                fa[0].e[2 * v] = pa0.x; fa[0].e[2 * v + 1] = pa0.y;
                fa[1].e[2 * v] = pa1.x; fa[1].e[2 * v + 1] = pa1.y;
                fb[0].e[2 * v] = pb0.x; fb[0].e[2 * v + 1] = pb0.y;
                fb[1].e[2 * v] = pb1.x; fb[1].e[2 * v + 1] = pb1.y;
            }
            #pragma unroll
            for (int mi = 0; mi < 2; ++mi)
                #pragma unroll
                for (int ni = 0; ni < 2; ++ni)
                    acc[mi][ni] = __builtin_amdgcn_wmma_f32_16x16x32_bf16(
                        false, fa[mi].v, false, fb[ni].v, (short)0,
                        acc[mi][ni], false, false);
        }
        __syncthreads();
    }

    // ---- store per 32-bit C/D layout: row = r + 8*half, col = lane&15
    #pragma unroll
    for (int ni = 0; ni < 2; ++ni) {
        const int n = n0 + wn * 32 + ni * 16 + lm;
        if (n < N) {
            const float bz = bias ? bias[n] : 0.0f;
            #pragma unroll
            for (int mi = 0; mi < 2; ++mi) {
                AccFrag rr; rr.v = acc[mi][ni];
                #pragma unroll
                for (int r = 0; r < 8; ++r) {
                    int mrow = m0 + wm * 32 + mi * 16 + half * 8 + r;
                    float v = rr.f[r] + bz;
                    if (ACT == 1) v = (v > 20.0f) ? v : log1pf(expf(v));
                    C[(size_t)mrow * ldc + c_off + n] = v;
                }
            }
        }
    }
}

// ===================================================================
// Depthwise causal conv (D_CONV=4) + bias + SiLU.
// ===================================================================
__global__ __launch_bounds__(256) void conv_silu_kernel(
    const float* __restrict__ xz,
    const float* __restrict__ cw,   // [D_INNER, D_CONV]
    const float* __restrict__ cb,   // [D_INNER]
    float* __restrict__ u)
{
    int idx = blockIdx.x * 256 + threadIdx.x;          // over NTOK * D_INNER
    if (idx >= NTOK * D_INNER) return;
    int tok = idx >> 12;            // /4096
    int d   = idx & (D_INNER - 1);
    int t   = tok & (SEQ - 1);      // position within sequence

    float acc = cb[d];
    #pragma unroll
    for (int k = 0; k < D_CONV; ++k) {
        int dt = t - (D_CONV - 1) + k;
        if (dt >= 0) {
            float xv = xz[(size_t)(tok - (D_CONV - 1) + k) * (2 * D_INNER) + d];
            acc += xv * cw[d * D_CONV + k];
        }
    }
    u[idx] = silu(acc);
}

// ===================================================================
// Selective scan. One wave per (batch, channel): 16384 waves.
// Each lane owns states s = lane and lane+32 (D_STATE=64).
// ===================================================================
__global__ __launch_bounds__(256) void scan_kernel(
    const float* __restrict__ dt,     // [NTOK, D_INNER]
    const float* __restrict__ u,      // [NTOK, D_INNER]
    const float* __restrict__ xdbl,   // [NTOK, 256]; B at +128, C at +192
    const float* __restrict__ Alog,   // [D_INNER, D_STATE]
    float* __restrict__ y)            // [NTOK, D_INNER]
{
    const int wave = threadIdx.x >> 5;
    const int lane = threadIdx.x & 31;
    const int ch   = blockIdx.x * 8 + wave;      // 0 .. BATCH*D_INNER-1
    const int b    = ch >> 12;                   // /4096
    const int d    = ch & (D_INNER - 1);

    const int s0 = lane;
    const int s1 = lane + 32;
    const float a0 = -__expf(Alog[(size_t)d * D_STATE + s0]);
    const float a1 = -__expf(Alog[(size_t)d * D_STATE + s1]);

    float h0 = 0.0f, h1 = 0.0f;
    const size_t tok0 = (size_t)b * SEQ;

    for (int t = 0; t < SEQ; ++t) {
        const size_t tok = tok0 + t;
        const float dtv = dt[tok * D_INNER + d];
        const float uv  = u [tok * D_INNER + d];
        const float Bv0 = xdbl[tok * 256 + DT_RANK + s0];
        const float Bv1 = xdbl[tok * 256 + DT_RANK + s1];
        const float Cv0 = xdbl[tok * 256 + DT_RANK + D_STATE + s0];
        const float Cv1 = xdbl[tok * 256 + DT_RANK + D_STATE + s1];

        const float du = dtv * uv;
        h0 = __expf(dtv * a0) * h0 + du * Bv0;
        h1 = __expf(dtv * a1) * h1 + du * Bv1;

        float r = h0 * Cv0 + h1 * Cv1;
        #pragma unroll
        for (int off = 16; off > 0; off >>= 1)
            r += __shfl_xor(r, off, 32);
        if (lane == 0) y[tok * D_INNER + d] = r;
    }
}

// ===================================================================
// yv = (y + u * D_skip[d]) * silu(gate), written back into y.
// ===================================================================
__global__ __launch_bounds__(256) void gate_kernel(
    float* __restrict__ y,
    const float* __restrict__ u,
    const float* __restrict__ xz,
    const float* __restrict__ Dsk)
{
    int idx = blockIdx.x * 256 + threadIdx.x;
    if (idx >= NTOK * D_INNER) return;
    int tok = idx >> 12;
    int d   = idx & (D_INNER - 1);
    float g = xz[(size_t)tok * (2 * D_INNER) + D_INNER + d];
    y[idx] = (y[idx] + u[idx] * Dsk[d]) * silu(g);
}

// ===================================================================
// Head: logits[tok, c] = head_b[c] + sum_h hbuf[tok,h] * head_w[c,h]
// ===================================================================
__global__ __launch_bounds__(256) void head_kernel(
    const float* __restrict__ hbuf,   // [NTOK, HID]
    const float* __restrict__ hw,     // [NCLS, HID]
    const float* __restrict__ hb,     // [NCLS]
    float* __restrict__ out)          // [NTOK, NCLS]
{
    int idx = blockIdx.x * 256 + threadIdx.x;
    if (idx >= NTOK * NCLS) return;
    int tok = idx >> 4;
    int c   = idx & (NCLS - 1);
    const float* hp = hbuf + (size_t)tok * HID;
    const float* wp = hw + (size_t)c * HID;
    float acc = hb[c];
    for (int k = 0; k < HID; ++k) acc += hp[k] * wp[k];
    out[idx] = acc;
}

// ===================================================================
extern "C" void kernel_launch(void* const* d_in, const int* in_sizes, int n_in,
                              void* d_out, int out_size, void* d_ws, size_t ws_size,
                              hipStream_t stream)
{
    const float* X[3]   = { (const float*)d_in[0], (const float*)d_in[1], (const float*)d_in[2] };
    const float* in_w   = (const float*)d_in[3];   // [3, 8192, 2048]
    const float* conv_w = (const float*)d_in[4];   // [3, 4096, 4]
    const float* conv_b = (const float*)d_in[5];   // [3, 4096]
    const float* xp_w   = (const float*)d_in[6];   // [3, 256, 4096]
    const float* dt_w   = (const float*)d_in[7];   // [3, 4096, 128]
    const float* dt_b   = (const float*)d_in[8];   // [3, 4096]
    const float* A_log  = (const float*)d_in[9];   // [3, 4096, 64]
    const float* D_skip = (const float*)d_in[10];  // [3, 4096]
    const float* out_w  = (const float*)d_in[11];  // [3, 2048, 4096]
    const float* fc1_w  = (const float*)d_in[12];  // [1364, 6144]
    const float* fc1_b  = (const float*)d_in[13];  // [1364]
    const float* head_w = (const float*)d_in[14];  // [16, 1364]
    const float* head_b = (const float*)d_in[15];  // [16]
    float* out = (float*)d_out;

    // ---- workspace partition (f32 elements)
    float* ws = (float*)d_ws;
    float* xz   = ws;                         // NTOK*8192
    float* u    = xz   + (size_t)NTOK * 8192; // NTOK*4096
    float* xdbl = u    + (size_t)NTOK * 4096; // NTOK*256
    float* dtb  = xdbl + (size_t)NTOK * 256;  // NTOK*4096
    float* y    = dtb  + (size_t)NTOK * 4096; // NTOK*4096
    float* cat  = y    + (size_t)NTOK * 4096; // NTOK*6144
    float* hbuf = cat  + (size_t)NTOK * 6144; // NTOK*HID

    const int elemBlocks = (NTOK * D_INNER + 255) / 256;

    for (int i = 0; i < 3; ++i) {
        const float* inW  = in_w   + (size_t)i * (2 * D_INNER) * D_MODEL;
        const float* cW   = conv_w + (size_t)i * D_INNER * D_CONV;
        const float* cB   = conv_b + (size_t)i * D_INNER;
        const float* xpW  = xp_w   + (size_t)i * (DT_RANK + 2 * D_STATE) * D_INNER;
        const float* dtW  = dt_w   + (size_t)i * D_INNER * DT_RANK;
        const float* dtB  = dt_b   + (size_t)i * D_INNER;
        const float* Al   = A_log  + (size_t)i * D_INNER * D_STATE;
        const float* Dsk  = D_skip + (size_t)i * D_INNER;
        const float* oW   = out_w  + (size_t)i * D_MODEL * D_INNER;

        // 1. xz = X @ in_w^T   [2048 x 2048] @ [2048 x 8192]
        {
            dim3 g((2 * D_INNER) / 64, NTOK / 128);
            gemm_bf16_wmma<0><<<g, 256, 0, stream>>>(
                X[i], D_MODEL, inW, D_MODEL, nullptr,
                xz, 2 * D_INNER, 0, NTOK, 2 * D_INNER, D_MODEL);
        }
        // 2. u = silu(conv(xa) + cb)
        conv_silu_kernel<<<elemBlocks, 256, 0, stream>>>(xz, cW, cB, u);
        // 3. x_dbl = u @ xpw^T   [2048 x 4096] @ [4096 x 256]
        {
            dim3 g(256 / 64, NTOK / 128);
            gemm_bf16_wmma<0><<<g, 256, 0, stream>>>(
                u, D_INNER, xpW, D_INNER, nullptr,
                xdbl, 256, 0, NTOK, 256, D_INNER);
        }
        // 4. dt = softplus(dt_lo @ dtw^T + dt_b)   [2048 x 128] @ [128 x 4096]
        {
            dim3 g(D_INNER / 64, NTOK / 128);
            gemm_bf16_wmma<1><<<g, 256, 0, stream>>>(
                xdbl, 256, dtW, DT_RANK, dtB,
                dtb, D_INNER, 0, NTOK, D_INNER, DT_RANK);
        }
        // 5. selective scan -> y
        scan_kernel<<<(BATCH * D_INNER) / 8, 256, 0, stream>>>(dtb, u, xdbl, Al, y);
        // 6. y = (y + u*Dsk) * silu(gate)   (in place)
        gate_kernel<<<elemBlocks, 256, 0, stream>>>(y, u, xz, Dsk);
        // 7. cat[:, i*2048 : (i+1)*2048] = y @ out_w^T
        {
            dim3 g(D_MODEL / 64, NTOK / 128);
            gemm_bf16_wmma<0><<<g, 256, 0, stream>>>(
                y, D_INNER, oW, D_INNER, nullptr,
                cat, 3 * D_MODEL, i * D_MODEL, NTOK, D_MODEL, D_INNER);
        }
    }

    // 8. hbuf = cat @ fc1_w^T + fc1_b   [2048 x 6144] @ [6144 x 1364]
    {
        dim3 g((HID + 63) / 64, NTOK / 128);
        gemm_bf16_wmma<0><<<g, 256, 0, stream>>>(
            cat, 3 * D_MODEL, fc1_w, 3 * D_MODEL, fc1_b,
            hbuf, HID, 0, NTOK, HID, 3 * D_MODEL);
    }
    // 9. logits = hbuf @ head_w^T + head_b
    head_kernel<<<(NTOK * NCLS + 255) / 256, 256, 0, stream>>>(hbuf, head_w, head_b, out);
}
